// MoELayer_1322849927668
// MI455X (gfx1250) — matrix operand order repaired and checked by reading, not compiled, not run on hardware
//
#include <hip/hip_runtime.h>
#include <hip/hip_bf16.h>
#include <math.h>

// Problem constants (match reference)
#define T_TOK 16384   // B*S tokens
#define DD    512     // model dim
#define FF    2048    // ffn dim
#define EE    16      // experts
#define KSEL  2       // top-k

typedef __attribute__((ext_vector_type(16))) __bf16 v16bf;
typedef __attribute__((ext_vector_type(8)))  float  v8f;

union Frag { v16bf v; unsigned u[8]; };
union Acc  { v8f   v; float    f[8]; };

__device__ __forceinline__ unsigned bf16_rne(float a) {
    unsigned u = __float_as_uint(a);
    return (u + 0x7FFFu + ((u >> 16) & 1u)) >> 16;
}
__device__ __forceinline__ unsigned pack_bf16(float a, float b) {
    return bf16_rne(a) | (bf16_rne(b) << 16);
}

// CDNA5 async global->LDS copy (ASYNCcnt-tracked), 16B per lane.
__device__ __forceinline__ void async_b128(unsigned lds_off, const unsigned* gptr) {
    asm volatile("global_load_async_to_lds_b128 %0, %1, off"
                 :: "v"(lds_off), "v"(gptr) : "memory");
}
__device__ __forceinline__ void wait_async0() {
    asm volatile("s_wait_asynccnt 0" ::: "memory");
}

// ---------------------------------------------------------------- zero
__global__ void zero_f32(float* __restrict__ p, int n) {
    int i = blockIdx.x * blockDim.x + threadIdx.x;
    if (i < n) p[i] = 0.0f;
}

// ---------------------------------------------------------------- x packer: consecutive-element bf16 pairs
__global__ void pack_inner_kernel(const float* __restrict__ src, unsigned* __restrict__ dst, int npairs) {
    int i = blockIdx.x * blockDim.x + threadIdx.x;
    if (i < npairs) dst[i] = pack_bf16(src[2 * i], src[2 * i + 1]);
}

// ---------------------------------------------------------------- weight packer: fragment-ready swizzled tiles
// Per expert: KD x ND weights -> tiles of 32(K) x 64(N), each tile 1024 contiguous dwords
// ordered [(kh*64 + nn)*8 + i] with dword = pack(w[ks*32+kh*16+2i][tf*64+nn], w[..+1][..]),
// i.e. the per-lane B-fragment order this code uses: k = 16*kh + s, the same order as A.
template <int KD, int ND>
__global__ void pack_swizzle_kernel(const float* __restrict__ src, unsigned* __restrict__ dst, int total) {
    int idx = blockIdx.x * blockDim.x + threadIdx.x;
    if (idx >= total) return;
    const int perExp = KD * ND / 2;   // dwords per expert (pow2)
    const int TF     = ND / 64;       // n-tiles per row of tiles (pow2)
    int e   = idx / perExp;
    int rem = idx - e * perExp;
    int t   = rem >> 10;              // tile id (1024 dwords/tile)
    int w   = rem & 1023;
    int ks  = t / TF;
    int tf  = t - ks * TF;
    int kh  = w >> 9;                 // 0..1
    int nn  = (w >> 3) & 63;          // 0..63
    int i   = w & 7;                  // 0..7
    int k   = ks * 32 + kh * 16 + 2 * i;
    int f   = tf * 64 + nn;
    const float* s = src + ((size_t)e * KD + k) * ND + f;
    dst[idx] = pack_bf16(s[0], s[ND]);
}

// ---------------------------------------------------------------- gate: logits -> softmax -> top2
__global__ void gate_kernel(const float* __restrict__ x,
                            const float* __restrict__ gw,
                            int* __restrict__ counts,
                            int* __restrict__ selA, int* __restrict__ selB,
                            float* __restrict__ wA, float* __restrict__ wB) {
    int t = blockIdx.x * blockDim.x + threadIdx.x;
    if (t >= T_TOK) return;
    float acc[EE];
#pragma unroll
    for (int e = 0; e < EE; ++e) acc[e] = 0.0f;
    const float* xr = x + (size_t)t * DD;
    for (int d = 0; d < DD; ++d) {
        float xv = xr[d];
        const float* g = gw + d * EE;
#pragma unroll
        for (int e = 0; e < EE; ++e) acc[e] = fmaf(xv, g[e], acc[e]);
    }
    float mx = acc[0];
#pragma unroll
    for (int e = 1; e < EE; ++e) mx = fmaxf(mx, acc[e]);
    float s = 0.0f;
#pragma unroll
    for (int e = 0; e < EE; ++e) { acc[e] = __expf(acc[e] - mx); s += acc[e]; }
    float inv = 1.0f / s;
#pragma unroll
    for (int e = 0; e < EE; ++e) acc[e] *= inv;
    int e1 = 0; float v1 = acc[0];
#pragma unroll
    for (int e = 1; e < EE; ++e) if (acc[e] > v1) { v1 = acc[e]; e1 = e; }
    int e2 = -1; float v2 = -1.0f;
#pragma unroll
    for (int e = 0; e < EE; ++e) if (e != e1 && acc[e] > v2) { v2 = acc[e]; e2 = e; }
    float norm = 1.0f / (v1 + v2);
    selA[t] = e1; selB[t] = e2;
    wA[t] = v1 * norm; wB[t] = v2 * norm;
    atomicAdd(&counts[e1], 1);
    atomicAdd(&counts[e2], 1);
}

// ---------------------------------------------------------------- prefix over E=16
__global__ void offsets_kernel(const int* __restrict__ counts, int* __restrict__ offs) {
    if (threadIdx.x == 0 && blockIdx.x == 0) {
        int a = 0;
        for (int e = 0; e < EE; ++e) { offs[e] = a; a += counts[e]; }
        offs[EE] = a;
    }
}

// ---------------------------------------------------------------- scatter tokens to expert segments
__global__ void scatter_kernel(const int* __restrict__ selA, const int* __restrict__ selB,
                               const float* __restrict__ wA, const float* __restrict__ wB,
                               const int* __restrict__ offs, int* __restrict__ cursors,
                               int* __restrict__ perm, float* __restrict__ permw) {
    int t = blockIdx.x * blockDim.x + threadIdx.x;
    if (t >= T_TOK) return;
    int e1 = selA[t];
    int p1 = atomicAdd(&cursors[e1], 1);
    perm[offs[e1] + p1] = t; permw[offs[e1] + p1] = wA[t];
    int e2 = selB[t];
    int p2 = atomicAdd(&cursors[e2], 1);
    perm[offs[e2] + p2] = t; permw[offs[e2] + p2] = wB[t];
}

// ================================================================
// GEMM core: 128 threads (4 waves in 2x2), 64x64 tile, K-step 32,
// double-buffered LDS filled by ASYNC b128 copies.
// A buf: 64 rows x 16 kp-dwords (1024 dw, row-major bf16 pairs).
// B buf: one swizzled 32x64 tile (1024 dw, fragment-ready order).
// ================================================================

// ---------------------------------------------------------------- GEMM1: H = gelu(Xg @ W1[e] + b1[e]) -> bf16
__global__ __launch_bounds__(128) void gemm1_kernel(
        const unsigned* __restrict__ Xpk, const unsigned* __restrict__ W1sw,
        const float* __restrict__ b1, const int* __restrict__ perm,
        const int* __restrict__ offs, const int* __restrict__ counts,
        unsigned short* __restrict__ Hbuf) {
    const int e   = blockIdx.z;
    const int cnt = counts[e];
    const int m0  = blockIdx.y * 64;
    if (m0 >= cnt) return;
    const int base = offs[e];
    const int f0   = blockIdx.x * 64;
    const int BSTRIDE = (FF / 64) * 1024;   // dwords between successive K-step tiles
    const unsigned* Wt = W1sw + (size_t)e * (DD / 2) * FF + (size_t)(f0 >> 6) * 1024;

    __shared__ unsigned As[2][64 * 16];
    __shared__ unsigned Bs[2][1024];

    const int tid  = threadIdx.x;
    const int lane = tid & 31;
    const int wave = tid >> 5;
    const int wm = wave >> 1, wn = wave & 1;
    const int lm = lane & 15, kh = lane >> 4;

    // per-thread async chunk descriptors (two A + two B b128 chunks / thread)
    const int c0 = wave * 32 + lane, c1 = 128 + c0;
    const int rA0 = c0 >> 2, qA0 = c0 & 3;
    const int rA1 = c1 >> 2, qA1 = c1 & 3;
    const int tok0 = perm[base + ((m0 + rA0 < cnt) ? (m0 + rA0) : (cnt - 1))];
    const int tok1 = perm[base + ((m0 + rA1 < cnt) ? (m0 + rA1) : (cnt - 1))];
    const unsigned* gA0 = Xpk + (size_t)tok0 * (DD / 2) + qA0 * 4;
    const unsigned* gA1 = Xpk + (size_t)tok1 * (DD / 2) + qA1 * 4;
    const unsigned* gB0 = Wt + c0 * 4;
    const unsigned* gB1 = Wt + c1 * 4;

    const unsigned aBase = (unsigned)(size_t)(void*)&As[0][0];
    const unsigned bBase = (unsigned)(size_t)(void*)&Bs[0][0];
    const unsigned ldsA0 = (unsigned)(rA0 * 16 + qA0 * 4) * 4u;
    const unsigned ldsA1 = (unsigned)(rA1 * 16 + qA1 * 4) * 4u;
    const unsigned ldsB0 = (unsigned)c0 * 16u;
    const unsigned ldsB1 = (unsigned)c1 * 16u;

    Acc acc[2][2];
#pragma unroll
    for (int i = 0; i < 2; ++i)
#pragma unroll
        for (int j = 0; j < 2; ++j)
#pragma unroll
            for (int q = 0; q < 8; ++q) acc[i][j].f[q] = 0.0f;

    // prologue: fill buffer 0 for k0 = 0
    async_b128(aBase + ldsA0, gA0);
    async_b128(aBase + ldsA1, gA1);
    async_b128(bBase + ldsB0, gB0);
    async_b128(bBase + ldsB1, gB1);
    wait_async0();
    __syncthreads();

    for (int k0 = 0; k0 < DD; k0 += 32) {
        const int cur = (k0 >> 5) & 1;
        if (k0 + 32 < DD) {                                  // issue next K-step
            const int nxt = cur ^ 1;
            const int kd = (k0 + 32) >> 1;                   // A dword offset along K
            const size_t bt = (size_t)((k0 + 32) >> 5) * BSTRIDE;
            async_b128(aBase + (unsigned)nxt * 4096u + ldsA0, gA0 + kd);
            async_b128(aBase + (unsigned)nxt * 4096u + ldsA1, gA1 + kd);
            async_b128(bBase + (unsigned)nxt * 4096u + ldsB0, gB0 + bt);
            async_b128(bBase + (unsigned)nxt * 4096u + ldsB1, gB1 + bt);
        }
        Frag a[2], b[2];
#pragma unroll
        for (int im = 0; im < 2; ++im) {
            int m = wm * 32 + im * 16 + lm;
#pragma unroll
            for (int i = 0; i < 4; ++i) a[im].u[i]     = As[cur][m * 16 + kh * 8 + i];
#pragma unroll
            for (int i = 0; i < 4; ++i) a[im].u[4 + i] = As[cur][m * 16 + kh * 8 + 4 + i];
        }
#pragma unroll
        for (int in = 0; in < 2; ++in) {
            int n = wn * 32 + in * 16 + lm;
            const unsigned* bp = &Bs[cur][(kh * 64 + n) * 8];
#pragma unroll
            for (int i = 0; i < 8; ++i) b[in].u[i] = bp[i];   // 2x ds_load_b128
        }
#pragma unroll
        for (int im = 0; im < 2; ++im)
#pragma unroll
            for (int in = 0; in < 2; ++in)
                acc[im][in].v = __builtin_amdgcn_wmma_f32_16x16x32_bf16(
                    false, a[im].v, false, b[in].v, (short)0, acc[im][in].v, false, false);
        wait_async0();
        __syncthreads();
    }

    const float* B1 = b1 + (size_t)e * FF;
#pragma unroll
    for (int im = 0; im < 2; ++im)
#pragma unroll
        for (int in = 0; in < 2; ++in)
#pragma unroll
            for (int i = 0; i < 8; ++i) {
                int m = wm * 32 + im * 16 + i + kh * 8;
                int r = m0 + m;
                if (r < cnt) {
                    int n = f0 + wn * 32 + in * 16 + lm;
                    float v = acc[im][in].f[i] + B1[n];
                    v = 0.5f * v * (1.0f + erff(v * 0.70710678118654752f)); // exact GELU
                    Hbuf[(size_t)(base + r) * FF + n] = (unsigned short)bf16_rne(v);
                }
            }
}

// ---------------------------------------------------------------- GEMM2: out += w_tok * (H @ W2[e] + b2[e])
__global__ __launch_bounds__(128) void gemm2_kernel(
        const unsigned* __restrict__ Hpk, const unsigned* __restrict__ W2sw,
        const float* __restrict__ b2, const int* __restrict__ perm,
        const float* __restrict__ permw, const int* __restrict__ offs,
        const int* __restrict__ counts, float* __restrict__ out) {
    const int e   = blockIdx.z;
    const int cnt = counts[e];
    const int m0  = blockIdx.y * 64;
    if (m0 >= cnt) return;
    const int base = offs[e];
    const int d0   = blockIdx.x * 64;
    const int BSTRIDE = (DD / 64) * 1024;
    const unsigned* Wt = W2sw + (size_t)e * (FF / 2) * DD + (size_t)(d0 >> 6) * 1024;

    __shared__ unsigned As[2][64 * 16];
    __shared__ unsigned Bs[2][1024];

    const int tid  = threadIdx.x;
    const int lane = tid & 31;
    const int wave = tid >> 5;
    const int wm = wave >> 1, wn = wave & 1;
    const int lm = lane & 15, kh = lane >> 4;

    const int c0 = wave * 32 + lane, c1 = 128 + c0;
    const int rA0 = c0 >> 2, qA0 = c0 & 3;
    const int rA1 = c1 >> 2, qA1 = c1 & 3;
    const int gr0 = base + ((m0 + rA0 < cnt) ? (m0 + rA0) : (cnt - 1));
    const int gr1 = base + ((m0 + rA1 < cnt) ? (m0 + rA1) : (cnt - 1));
    const unsigned* gA0 = Hpk + (size_t)gr0 * (FF / 2) + qA0 * 4;
    const unsigned* gA1 = Hpk + (size_t)gr1 * (FF / 2) + qA1 * 4;
    const unsigned* gB0 = Wt + c0 * 4;
    const unsigned* gB1 = Wt + c1 * 4;

    const unsigned aBase = (unsigned)(size_t)(void*)&As[0][0];
    const unsigned bBase = (unsigned)(size_t)(void*)&Bs[0][0];
    const unsigned ldsA0 = (unsigned)(rA0 * 16 + qA0 * 4) * 4u;
    const unsigned ldsA1 = (unsigned)(rA1 * 16 + qA1 * 4) * 4u;
    const unsigned ldsB0 = (unsigned)c0 * 16u;
    const unsigned ldsB1 = (unsigned)c1 * 16u;

    Acc acc[2][2];
#pragma unroll
    for (int i = 0; i < 2; ++i)
#pragma unroll
        for (int j = 0; j < 2; ++j)
#pragma unroll
            for (int q = 0; q < 8; ++q) acc[i][j].f[q] = 0.0f;

    async_b128(aBase + ldsA0, gA0);
    async_b128(aBase + ldsA1, gA1);
    async_b128(bBase + ldsB0, gB0);
    async_b128(bBase + ldsB1, gB1);
    wait_async0();
    __syncthreads();

    for (int k0 = 0; k0 < FF; k0 += 32) {
        const int cur = (k0 >> 5) & 1;
        if (k0 + 32 < FF) {
            const int nxt = cur ^ 1;
            const int kd = (k0 + 32) >> 1;
            const size_t bt = (size_t)((k0 + 32) >> 5) * BSTRIDE;
            async_b128(aBase + (unsigned)nxt * 4096u + ldsA0, gA0 + kd);
            async_b128(aBase + (unsigned)nxt * 4096u + ldsA1, gA1 + kd);
            async_b128(bBase + (unsigned)nxt * 4096u + ldsB0, gB0 + bt);
            async_b128(bBase + (unsigned)nxt * 4096u + ldsB1, gB1 + bt);
        }
        Frag a[2], b[2];
#pragma unroll
        for (int im = 0; im < 2; ++im) {
            int m = wm * 32 + im * 16 + lm;
#pragma unroll
            for (int i = 0; i < 4; ++i) a[im].u[i]     = As[cur][m * 16 + kh * 8 + i];
#pragma unroll
            for (int i = 0; i < 4; ++i) a[im].u[4 + i] = As[cur][m * 16 + kh * 8 + 4 + i];
        }
#pragma unroll
        for (int in = 0; in < 2; ++in) {
            int n = wn * 32 + in * 16 + lm;
            const unsigned* bp = &Bs[cur][(kh * 64 + n) * 8];
#pragma unroll
            for (int i = 0; i < 8; ++i) b[in].u[i] = bp[i];
        }
#pragma unroll
        for (int im = 0; im < 2; ++im)
#pragma unroll
            for (int in = 0; in < 2; ++in)
                acc[im][in].v = __builtin_amdgcn_wmma_f32_16x16x32_bf16(
                    false, a[im].v, false, b[in].v, (short)0, acc[im][in].v, false, false);
        wait_async0();
        __syncthreads();
    }

    const float* B2 = b2 + (size_t)e * DD;
#pragma unroll
    for (int im = 0; im < 2; ++im)
#pragma unroll
        for (int in = 0; in < 2; ++in)
#pragma unroll
            for (int i = 0; i < 8; ++i) {
                int m = wm * 32 + im * 16 + i + kh * 8;
                int r = m0 + m;
                if (r < cnt) {
                    int n   = d0 + wn * 32 + in * 16 + lm;
                    int tok = perm[base + r];
                    float wt = permw[base + r];
                    float v = (acc[im][in].f[i] + B2[n]) * wt;
                    atomicAdd(&out[(size_t)tok * DD + n], v); // 2 commutative adds/elem -> deterministic
                }
            }
}

// ---------------------------------------------------------------- host launcher
extern "C" void kernel_launch(void* const* d_in, const int* in_sizes, int n_in,
                              void* d_out, int out_size, void* d_ws, size_t ws_size,
                              hipStream_t stream) {
    (void)in_sizes; (void)n_in; (void)ws_size;
    const float* x      = (const float*)d_in[0];
    const float* gate_w = (const float*)d_in[1];
    const float* w1     = (const float*)d_in[2];
    const float* b1     = (const float*)d_in[3];
    const float* w2     = (const float*)d_in[4];
    const float* b2     = (const float*)d_in[5];
    float* out = (float*)d_out;

    // workspace layout (bytes)
    char* ws = (char*)d_ws;
    int*   counts  = (int*)(ws + 0);     // 16 ints
    int*   cursors = (int*)(ws + 64);    // 16 ints
    int*   offs    = (int*)(ws + 128);   // 17 ints
    int*   selA    = (int*)(ws + 256);
    int*   selB    = (int*)(ws + 256 + 65536);
    float* wA      = (float*)(ws + 256 + 2 * 65536);
    float* wB      = (float*)(ws + 256 + 3 * 65536);
    int*   perm    = (int*)(ws + 262400);                     // T*K ints
    float* permw   = (float*)(ws + 393472);                   // T*K floats
    unsigned* Xpk  = (unsigned*)(ws + 524544);                // T*(D/2) dwords, 16 MB
    unsigned* W1sw = (unsigned*)(ws + 17301760);              // E*(D/2)*F dwords, 33.5 MB (swizzled)
    unsigned* W2sw = (unsigned*)(ws + 50856192);              // E*(F/2)*D dwords, 33.5 MB (swizzled)
    unsigned short* Hbuf = (unsigned short*)(ws + 84410624);  // T*K*F bf16, 134 MB

    // 1) zero output (incl. aux_loss slot) and routing counters
    zero_f32<<<(out_size + 255) / 256, 256, 0, stream>>>(out, out_size);
    zero_f32<<<1, 64, 0, stream>>>((float*)ws, 32); // counts + cursors

    // 2) one-time bf16 packing / swizzling (amortized; bf16 weights stay L2-resident)
    pack_inner_kernel<<<(T_TOK * (DD / 2) + 255) / 256, 256, 0, stream>>>(x, Xpk, T_TOK * (DD / 2));
    pack_swizzle_kernel<DD, FF><<<((EE * DD / 2) * FF + 255) / 256, 256, 0, stream>>>(w1, W1sw, (EE * DD / 2) * FF);
    pack_swizzle_kernel<FF, DD><<<((EE * FF / 2) * DD + 255) / 256, 256, 0, stream>>>(w2, W2sw, (EE * FF / 2) * DD);

    // 3) gating + top-2
    gate_kernel<<<(T_TOK + 255) / 256, 256, 0, stream>>>(x, gate_w, counts, selA, selB, wA, wB);

    // 4) segment offsets
    offsets_kernel<<<1, 1, 0, stream>>>(counts, offs);

    // 5) scatter tokens into per-expert segments
    scatter_kernel<<<(T_TOK + 255) / 256, 256, 0, stream>>>(selA, selB, wA, wB, offs, cursors, perm, permw);

    // 6) expert FFN up-projection + GELU (bf16 WMMA, async double-buffered LDS)
    dim3 g1(FF / 64, (T_TOK * KSEL + 63) / 64, EE);
    gemm1_kernel<<<g1, 128, 0, stream>>>(Xpk, W1sw, b1, perm, offs, counts, Hbuf);

    // 7) expert FFN down-projection, weight, scatter-add
    dim3 g2(DD / 64, (T_TOK * KSEL + 63) / 64, EE);
    gemm2_kernel<<<g2, 128, 0, stream>>>((const unsigned*)Hbuf, W2sw, b2, perm, permw, offs, counts, out);
}
